// LSTMModel_67954972557360
// MI455X (gfx1250) — matrix operand order repaired and checked by reading, not compiled, not run on hardware
//
#include <hip/hip_runtime.h>
#include <hip/hip_bf16.h>

typedef _Float16 half_t;
typedef __attribute__((ext_vector_type(16))) _Float16 v16h;
typedef __attribute__((ext_vector_type(8)))  _Float16 v8h;
typedef __attribute__((ext_vector_type(2)))  _Float16 v2h;
typedef __attribute__((ext_vector_type(8)))  float    v8f;

#define B_     4096
#define T_     256
#define H_     50
#define NG     200    // 4*H gate outputs
#define NPAD2  224    // 14 N-tiles of 16 (cols >= 200 are zero padding)
#define KMAX   128    // max padded K across layers (layer1: 64)
#define WBYT   57344  // max(NPAD2*KMAX*2, 16*NPAD2*4): W image / gate scratch

// Branchless fast transcendentals (saturate correctly at +/-inf):
__device__ __forceinline__ float fast_tanh(float x) {
    float e = __expf(2.0f * x);
    return 1.0f - 2.0f * __builtin_amdgcn_rcpf(e + 1.0f);
}
__device__ __forceinline__ float fast_sig(float x) {
    return __builtin_amdgcn_rcpf(1.0f + __expf(-x));
}

// One-shot input repack: x[B][T][13] f32  ->  Xc[T][B][14] f16 (col 13 = 0).
__global__ __launch_bounds__(256) void xprep_kernel(
    const float* __restrict__ x, half_t* __restrict__ Xc)
{
    size_t idx = (size_t)blockIdx.x * 256 + threadIdx.x;   // over T*B*14
    if (idx >= (size_t)T_ * B_ * 14) return;
    int f = (int)(idx % 14);
    size_t bt = idx / 14;
    int b = (int)(bt % B_);
    int t = (int)(bt / B_);
    float v = (f < 13) ? x[((size_t)b * T_ + t) * 13 + f] : 0.0f;
    Xc[idx] = (half_t)v;
}

// Per-wave persistent T-loop. Wave owns N-tiles [N0,N0+NT) and the matching
// gate columns (tile-aligned) -> gS is wave-private (same-wave in-order LDS),
// one barrier per step. A operand double-buffered by step parity.
template<int N0, int NT, int J0, int PJ, int NITER,
         int FINP, int KTOT, bool WRITE_Y>
__device__ __forceinline__ void lstm_loop(
    half_t* __restrict__ Wlds, float* __restrict__ biasLds,
    float*  __restrict__ gS,   half_t* __restrict__ Alds,
    const uint32_t* __restrict__ Xd, half_t* __restrict__ Y,
    int tid, int lane, int bbase)
{
    constexpr int KSTEPS = KTOT / 32;
    constexpr int FD     = FINP / 2;          // dwords per input row
    constexpr int NPFD   = 8 * FINP;          // dwords per 16-row step segment
    constexpr int NPF    = (NPFD + 127) / 128;

    // preload B fragments + bias into registers (loop-invariant)
    v16h bfr[NT][KSTEPS];
    float bvr[NT];
#pragma unroll
    for (int nt = 0; nt < NT; ++nt) {
        int ncol = (N0 + nt)*16 + (lane & 15);
        bvr[nt] = biasLds[ncol];
#pragma unroll
        for (int ks = 0; ks < KSTEPS; ++ks) {
            int kb = ks*32 + (lane >> 4)*8;
            v8h blo = *(const v8h*)&Wlds[ncol*KTOT + kb];
            v8h bhi = *(const v8h*)&Wlds[ncol*KTOT + kb + 16];
            bfr[nt][ks] = __builtin_shufflevector(blo, bhi,
                0,1,2,3,4,5,6,7,8,9,10,11,12,13,14,15);
        }
    }
    __syncthreads();   // W/bias reads done by all waves; gS (aliased) free

    float creg[2*NITER];
#pragma unroll
    for (int i = 0; i < 2*NITER; ++i) creg[i] = 0.0f;

    for (int t = 0; t < T_; ++t) {
        const int bufp = t & 1;
        const int bufq = bufp ^ 1;

        // prefetch x_{t+1} into registers
        uint32_t px[NPF];
        {
            int tp = (t + 1 < T_) ? t + 1 : t;
            size_t seg = ((size_t)tp * B_ + bbase) * FD;
#pragma unroll
            for (int k = 0; k < NPF; ++k) {
                int e = tid + k*128;
                px[k] = (e < NPFD) ? Xd[seg + e] : 0u;
            }
        }
        // A fragments (16x32 f16)
        v16h afrag[KSTEPS];
#pragma unroll
        for (int ks = 0; ks < KSTEPS; ++ks) {
            const half_t* ap = &Alds[(bufp*16 + (lane & 15))*KTOT];
            int c0 = ks*32 + (lane >> 4)*8;
            v8h lo = *(const v8h*)(ap + c0);
            v8h hi = *(const v8h*)(ap + c0 + 16);
            afrag[ks] = __builtin_shufflevector(lo, hi,
                0,1,2,3,4,5,6,7,8,9,10,11,12,13,14,15);
        }

        // WMMA GEMM: g = A @ W^T + bias (B register-resident)
#pragma unroll
        for (int nt = 0; nt < NT; ++nt) {
            int n = N0 + nt;
            float bv = bvr[nt];
            v8f acc = { bv, bv, bv, bv, bv, bv, bv, bv };
#pragma unroll
            for (int ks = 0; ks < KSTEPS; ++ks) {
                acc = __builtin_amdgcn_wmma_f32_16x16x32_f16(
                    false, afrag[ks], false, bfr[nt][ks], (short)0, acc,
                    false, false);
            }
#pragma unroll
            for (int r = 0; r < 8; ++r) {
                int m = r + ((lane >> 4) << 3);
                gS[m*NPAD2 + n*16 + (lane & 15)] = acc[r];
            }
        }

        // gates + state update on this wave's own gate columns
#pragma unroll
        for (int i = 0; i < NITER; ++i) {
            int q = lane + i * 32;
            if (NITER * 32 == 16 * PJ || q < 16 * PJ) {
                int m = q / PJ;
                int j = J0 + (q % PJ) * 2;
                const float* gp = &gS[m*NPAD2 + 4*j];
                float4 q0 = *(const float4*)(gp);
                float4 q1 = *(const float4*)(gp + 4);
                float c0 = fast_sig(q0.y) * creg[2*i]
                         + fast_sig(q0.x) * fast_tanh(q0.z);
                float c1 = fast_sig(q1.y) * creg[2*i+1]
                         + fast_sig(q1.x) * fast_tanh(q1.z);
                creg[2*i]   = c0;
                creg[2*i+1] = c1;
                float h0 = fast_sig(q0.w) * fast_tanh(c0);
                float h1 = fast_sig(q1.w) * fast_tanh(c1);
                v2h hp = { (half_t)h0, (half_t)h1 };
                *(v2h*)&Alds[(bufq*16 + m)*KTOT + FINP + j] = hp;
                if (WRITE_Y)
                    *(v2h*)&Y[((size_t)t*B_ + bbase + m)*H_ + j] = hp;
            }
        }

        // commit prefetched x_{t+1}
#pragma unroll
        for (int k = 0; k < NPF; ++k) {
            int e = tid + k*128;
            if (e < NPFD) {
                int r = e / FD, f2 = (e % FD) * 2;
                *(uint32_t*)&Alds[(bufq*16 + r)*KTOT + f2] = px[k];
            }
        }
        __syncthreads();   // single barrier per step
    }
}

// Cooperative per-layer setup: zero LDS, stage W (gate-interleaved cols,
// n' = 4j+gate) + bias into LDS, stage x_0 into A buffer 0.
template<int FIN, int FINP, int KTOT>
__device__ __forceinline__ void setup_layer(
    half_t* __restrict__ Wlds, float* __restrict__ biasLds,
    half_t* __restrict__ Alds,
    const float* __restrict__ Wih, const float* __restrict__ Whh,
    const float* __restrict__ bih, const float* __restrict__ bhh,
    const uint32_t* __restrict__ Xd, int tid, int bbase)
{
    constexpr int FD   = FINP / 2;
    constexpr int NPFD = 8 * FINP;
    constexpr int NPF  = (NPFD + 127) / 128;

    for (int i = tid; i < NPAD2*KTOT; i += 128) Wlds[i] = (half_t)0.0f;
    for (int i = tid; i < 2*16*KMAX; i += 128) Alds[i] = (half_t)0.0f;
    for (int i = tid; i < NPAD2; i += 128) biasLds[i] = 0.0f;
    __syncthreads();
    for (int i = tid; i < NG; i += 128) {
        int np = (i % H_) * 4 + (i / H_);
        biasLds[np] = bih[i] + bhh[i];
    }
    for (int i = tid; i < NG*FIN; i += 128) {
        int g = i / FIN, f = i % FIN;
        int np = (g % H_) * 4 + (g / H_);
        Wlds[np*KTOT + f] = (half_t)Wih[i];
    }
    for (int i = tid; i < NG*H_; i += 128) {
        int g = i / H_, j = i % H_;
        int np = (g % H_) * 4 + (g / H_);
        Wlds[np*KTOT + FINP + j] = (half_t)Whh[i];
    }
    {
        size_t seg = (size_t)bbase * FD;    // t = 0
#pragma unroll
        for (int k = 0; k < NPF; ++k) {
            int e = tid + k*128;
            if (e < NPFD) {
                uint32_t v = Xd[seg + e];
                int r = e / FD, f2 = (e % FD) * 2;
                *(uint32_t*)&Alds[r*KTOT + f2] = v;
            }
        }
    }
    __syncthreads();
}

template<int FIN, int FINP, int KTOT, bool WRITE_Y>
__device__ __forceinline__ void run_layer(
    half_t* Wlds, float* biasLds, float* gS, half_t* Alds,
    const float* Wih, const float* Whh, const float* bih, const float* bhh,
    const half_t* Xin, half_t* Y, int tid, int lane, int wave, int bbase)
{
    const uint32_t* Xd = (const uint32_t*)Xin;
    setup_layer<FIN, FINP, KTOT>(Wlds, biasLds, Alds,
                                 Wih, Whh, bih, bhh, Xd, tid, bbase);
    // wave0: tiles 0-3 / j [0,16)    wave1: tiles 4-6  / j [16,28)
    // wave2: tiles 7-10 / j [28,44)  wave3: tiles 11-13 / j [44,50)
    switch (wave) {
    case 0: lstm_loop< 0,4, 0,8,4, FINP,KTOT,WRITE_Y>(
                Wlds, biasLds, gS, Alds, Xd, Y, tid, lane, bbase); break;
    case 1: lstm_loop< 4,3,16,6,3, FINP,KTOT,WRITE_Y>(
                Wlds, biasLds, gS, Alds, Xd, Y, tid, lane, bbase); break;
    case 2: lstm_loop< 7,4,28,8,4, FINP,KTOT,WRITE_Y>(
                Wlds, biasLds, gS, Alds, Xd, Y, tid, lane, bbase); break;
    default: lstm_loop<11,3,44,3,2, FINP,KTOT,WRITE_Y>(
                Wlds, biasLds, gS, Alds, Xd, Y, tid, lane, bbase); break;
    }
}

// Fused persistent kernel: 3 LSTM layers + FC head. No inter-WG dependency:
// each WG owns 16 batch rows end-to-end ([T][B][H] layout makes layer n+1
// consume exactly what this WG produced in layer n).
__global__ __launch_bounds__(128) void lstm_fused_kernel(
    const half_t* __restrict__ Xc,     // [T][B][14] f16
    const float* __restrict__ Wih1, const float* __restrict__ Whh1,
    const float* __restrict__ bih1, const float* __restrict__ bhh1,
    const float* __restrict__ Wih2, const float* __restrict__ Whh2,
    const float* __restrict__ bih2, const float* __restrict__ bhh2,
    const float* __restrict__ Wih3, const float* __restrict__ Whh3,
    const float* __restrict__ bih3, const float* __restrict__ bhh3,
    half_t* __restrict__ Y,            // [T][B][50] f16 (in-place L2/L3)
    const float* __restrict__ fc1_w, const float* __restrict__ fc1_b,
    const float* __restrict__ fc2_w, const float* __restrict__ fc2_b,
    const float* __restrict__ fc3_w, const float* __restrict__ fc3_b,
    float* __restrict__ out)           // [B][240]
{
    extern __shared__ char smem[];
    half_t* Wlds    = (half_t*)smem;                 // [NPAD2][KTOT] (setup)
    float*  gS      = (float*)smem;                  // [16][NPAD2] (aliases W)
    float*  biasLds = (float*)(smem + WBYT);         // [NPAD2]
    half_t* Alds    = (half_t*)(smem + WBYT + NPAD2*4); // [2 buf][16][KMAX]

    const int tid   = threadIdx.x;
    const int lane  = tid & 31;
    const int wave  = tid >> 5;
    const int bbase = blockIdx.x * 16;

    run_layer<13, 14, 64, true >(Wlds, biasLds, gS, Alds,
        Wih1, Whh1, bih1, bhh1, Xc, Y, tid, lane, wave, bbase);
    // make this WG's Y stores visible before reading them back (LDS barrier
    // alone does not cover global stores)
    asm volatile("s_wait_storecnt 0x0" ::: "memory");
    __syncthreads();
    run_layer<50, 50, 128, true >(Wlds, biasLds, gS, Alds,
        Wih2, Whh2, bih2, bhh2, Y, Y, tid, lane, wave, bbase);
    asm volatile("s_wait_storecnt 0x0" ::: "memory");
    __syncthreads();
    run_layer<50, 50, 128, false>(Wlds, biasLds, gS, Alds,
        Wih3, Whh3, bih3, bhh3, Y, Y, tid, lane, wave, bbase);

    // ---- FC head on this WG's 16 rows; h_final lives in A buffer 0 ----
    // region0 (W/gS, now dead): fc1_w | fc2_w | fc3_w | a2 | biases (f32)
    float* w1  = (float*)smem;            // 64*50
    float* w2  = w1 + 64*50;              // 32*64
    float* w3  = w2 + 32*64;              // 240*32
    float* a2s = w3 + 240*32;             // 16*32
    float* bx  = a2s + 16*32;             // 64 + 32 + 240
    float* a1s = (float*)((char*)Alds + 16*KMAX*2); // A buffer 1: 16*64 f32
    const half_t* hS = Alds;              // h[r][j] = Alds[r*KMAX + 50 + j]

    for (int i = tid; i < 64*50;  i += 128) w1[i] = fc1_w[i];
    for (int i = tid; i < 32*64;  i += 128) w2[i] = fc2_w[i];
    for (int i = tid; i < 240*32; i += 128) w3[i] = fc3_w[i];
    if (tid < 64) bx[tid]      = fc1_b[tid];
    if (tid < 32) bx[64 + tid] = fc2_b[tid];
    for (int i = tid; i < 240; i += 128) bx[96 + i] = fc3_b[i];
    __syncthreads();

    const int r  = tid >> 3;              // row 0..15
    const int ot = tid & 7;               // 8 threads per row
    {
        float hv[H_];
#pragma unroll
        for (int k = 0; k < H_; ++k) hv[k] = (float)hS[r*KMAX + 50 + k];
#pragma unroll
        for (int i = 0; i < 8; ++i) {
            int o = ot + 8*i;
            float s = bx[o];
#pragma unroll
            for (int k = 0; k < H_; ++k) s += hv[k] * w1[o*H_ + k];
            a1s[r*64 + o] = fmaxf(s, 0.0f);
        }
    }
    __syncthreads();
    {
        float av[64];
#pragma unroll
        for (int k = 0; k < 64; ++k) av[k] = a1s[r*64 + k];
#pragma unroll
        for (int i = 0; i < 4; ++i) {
            int o = ot + 8*i;
            float s = bx[64 + o];
#pragma unroll
            for (int k = 0; k < 64; ++k) s += av[k] * w2[o*64 + k];
            a2s[r*32 + o] = fmaxf(s, 0.0f);
        }
    }
    __syncthreads();
    {
        float av[32];
#pragma unroll
        for (int k = 0; k < 32; ++k) av[k] = a2s[r*32 + k];
        for (int i = 0; i < 30; ++i) {
            int o = ot + 8*i;
            float s = bx[96 + o];
#pragma unroll
            for (int k = 0; k < 32; ++k) s += av[k] * w3[o*32 + k];
            out[(size_t)(bbase + r)*240 + o] = s;
        }
    }
}

extern "C" void kernel_launch(void* const* d_in, const int* in_sizes, int n_in,
                              void* d_out, int out_size, void* d_ws, size_t ws_size,
                              hipStream_t stream) {
    const float* x     = (const float*)d_in[0];
    const float* Wih1  = (const float*)d_in[1];
    const float* Whh1  = (const float*)d_in[2];
    const float* bih1  = (const float*)d_in[3];
    const float* bhh1  = (const float*)d_in[4];
    const float* Wih2  = (const float*)d_in[5];
    const float* Whh2  = (const float*)d_in[6];
    const float* bih2  = (const float*)d_in[7];
    const float* bhh2  = (const float*)d_in[8];
    const float* Wih3  = (const float*)d_in[9];
    const float* Whh3  = (const float*)d_in[10];
    const float* bih3  = (const float*)d_in[11];
    const float* bhh3  = (const float*)d_in[12];
    const float* fc1_w = (const float*)d_in[13];
    const float* fc1_b = (const float*)d_in[14];
    const float* fc2_w = (const float*)d_in[15];
    const float* fc2_b = (const float*)d_in[16];
    const float* fc3_w = (const float*)d_in[17];
    const float* fc3_b = (const float*)d_in[18];

    // workspace: Xc [T][B][14] f16 | Y [T][B][50] f16 (in-place L2/L3)
    half_t* Xc  = (half_t*)d_ws;
    half_t* Yws = (half_t*)((char*)d_ws + (size_t)T_ * B_ * 14 * sizeof(half_t));

    const size_t ldsz = WBYT + NPAD2*4 + 2*16*KMAX*2;   // ~65 KB

    {
        size_t nprep = (size_t)T_ * B_ * 14;
        dim3 g((unsigned)((nprep + 255) / 256));
        xprep_kernel<<<g, dim3(256), 0, stream>>>(x, Xc);
    }
    lstm_fused_kernel<<<dim3(B_ / 16), dim3(128), ldsz, stream>>>(
        Xc, Wih1, Whh1, bih1, bhh1, Wih2, Whh2, bih2, bhh2,
        Wih3, Whh3, bih3, bhh3, Yws,
        fc1_w, fc1_b, fc2_w, fc2_b, fc3_w, fc3_b, (float*)d_out);
}
